// GTCN_TCN_10050223472915
// MI455X (gfx1250) — matrix operand order repaired and checked by reading, not compiled, not run on hardware
//
#include <hip/hip_runtime.h>
#include <hip/hip_bf16.h>
#include <stdint.h>
#include <math.h>

// ---------------------------------------------------------------------------
// Types for CDNA5 WMMA (wave32). bf16 operands, f32 accumulate.
// ---------------------------------------------------------------------------
typedef __bf16 bf16;
typedef __attribute__((ext_vector_type(16))) __bf16 v16bf;
typedef __attribute__((ext_vector_type(8)))  __bf16 v8bf;
typedef __attribute__((ext_vector_type(8)))  float  v8f;

#define NUM_NODES 30
#define LPAD 32          // pad L 30 -> 32 so 16-row WMMA tiles align per graph
#define HEADS 4
#define CH 64
#define HC 256
#define F_IN 64
#define BG 1024          // graphs
#define EPG 240
#define NN (BG * NUM_NODES)      // 30720 nodes
#define E0 (BG * EPG)            // 245760 edges
#define ETOT (E0 + NN)           // + self loops = 276480
#define OUT_ELEMS (BG * 144)     // 147456
#define ALPHA_ELEMS (ETOT * HEADS)

__device__ inline v8f wmma_bf16(v16bf a, v16bf b, v8f c) {
  return __builtin_amdgcn_wmma_f32_16x16x32_bf16(false, a, false, b, (short)0, c,
                                                 false, false);
}

__device__ inline v16bf zero16() {
  v16bf a;
#pragma unroll
  for (int i = 0; i < 16; i++) a[i] = (bf16)0.0f;
  return a;
}

// A fragment 16x32 (MxK) bf16. Lane l holds row (l&15); K element e maps to
// ((e>>3)<<4) + ((l>>4)<<3) + (e&7)  -> two contiguous 8-elem (16B) chunks.
// `p` must already point at (row, k0 + ((l>>4)<<3)).
__device__ inline v16bf load_a(const bf16* p) {
  union { v16bf v; v8bf h[2]; } u;
  u.h[0] = *(const v8bf*)(p);
  u.h[1] = *(const v8bf*)(p + 16);
  return u.v;
}

// B fragment 32x16 (KxN) bf16 from row-major weight W[N, K] (row = out col).
// Lane l holds column (l&15); K element e maps to
// ((e>>3)<<4) + ((l>>4)<<3) + (e&7) (same as A fragment).
__device__ inline v16bf load_b16(const bf16* __restrict__ W, int ldk, int n0, int k0) {
  int l = threadIdx.x & 31;
  const bf16* p = W + (size_t)(n0 + (l & 15)) * ldk + k0 + ((l >> 4) << 3);
  union { v16bf v; v8bf h[2]; } u;
  u.h[0] = *(const v8bf*)(p);
  u.h[1] = *(const v8bf*)(p + 16);
  return u.v;
}

// ---------------------------------------------------------------------------
// fp32 -> bf16 conversion kernels
// ---------------------------------------------------------------------------
__global__ void k_f2bf(const float* __restrict__ s, bf16* __restrict__ d, int n) {
  int i = blockIdx.x * blockDim.x + threadIdx.x;
  if (i < n) d[i] = (bf16)s[i];
}

// conv weights [Cout, Cin, 3] -> tap-major bf16 [3][Cout][Cin]
__global__ void k_conv_w(const float* __restrict__ s, bf16* __restrict__ d,
                         int cout, int cin) {
  int i = blockIdx.x * blockDim.x + threadIdx.x;
  int tot = cout * cin * 3;
  if (i >= tot) return;
  int k = i % 3;
  int ci = (i / 3) % cin;
  int co = i / (3 * cin);
  d[((size_t)k * cout + co) * cin + ci] = (bf16)s[i];
}

// ---------------------------------------------------------------------------
// GATv2 node transforms: xl = x @ Wl^T + bl, xr = x @ Wr^T + br  (fp32 out)
// M = 30720, K = 64, N = 256. grid: (1920 Mtiles, 2 col-halves, 2 matrices)
// Two independent WMMAs (no accumulator chain -> no hazard NOPs).
// ---------------------------------------------------------------------------
__global__ __launch_bounds__(256) void k_xform(
    const bf16* __restrict__ xbf, const bf16* __restrict__ Wl,
    const bf16* __restrict__ Wr, const float* __restrict__ bl,
    const float* __restrict__ br, float* __restrict__ xl, float* __restrict__ xr) {
  int l = threadIdx.x & 31;
  int w = threadIdx.x >> 5;
  int mt = blockIdx.x;
  int n0 = blockIdx.y * 128 + w * 16;
  const bf16* W = blockIdx.z ? Wr : Wl;
  const float* bias = blockIdx.z ? br : bl;
  float* out = blockIdx.z ? xr : xl;

  const bf16* arow =
      xbf + (size_t)(mt * 16 + (l & 15)) * F_IN + ((l >> 4) << 3);
  v8f acc0 = {}, acc1 = {};
  acc0 = wmma_bf16(load_a(arow), load_b16(W, F_IN, n0, 0), acc0);
  acc1 = wmma_bf16(load_a(arow + 32), load_b16(W, F_IN, n0, 32), acc1);
  v8f acc = acc0 + acc1;

  int col = n0 + (l & 15);
  float bv = bias[col];
#pragma unroll
  for (int i = 0; i < 8; i++) {
    int row = mt * 16 + i + ((l >> 4) << 3);
    out[(size_t)row * HC + col] = acc[i] + bv;
  }
}

// ---------------------------------------------------------------------------
// GATv2 edge attention + segment softmax + aggregation. One block per graph.
// Writes alpha (d_out region) and h [B, 32, 256] (fp32 + bf16, rows 30/31 = 0).
// ---------------------------------------------------------------------------
__global__ __launch_bounds__(256) void k_gat(
    const float* __restrict__ xl, const float* __restrict__ xr,
    const int* __restrict__ ei, const float* __restrict__ att,
    const float* __restrict__ gbias, float* __restrict__ alpha_out,
    float* __restrict__ h_f32, bf16* __restrict__ h_bf) {
  __shared__ float xls[NUM_NODES * HC];
  __shared__ float xrs[NUM_NODES * HC];
  __shared__ float aggs[NUM_NODES * HC];
  __shared__ float es[(EPG + NUM_NODES) * HEADS];
  __shared__ float ms[NUM_NODES * HEADS];
  __shared__ float ss[NUM_NODES * HEADS];
  __shared__ int srcs[EPG + NUM_NODES];
  __shared__ int dsts[EPG + NUM_NODES];
  __shared__ float atts[HEADS * CH];

  const int b = blockIdx.x;
  const int tid = threadIdx.x;
  const int EL = EPG + NUM_NODES;  // 270 incl. self loops
  const size_t nb = (size_t)b * NUM_NODES;

  for (int i = tid; i < NUM_NODES * HC; i += 256) {
    xls[i] = xl[nb * HC + i];
    xrs[i] = xr[nb * HC + i];
    aggs[i] = 0.f;
  }
  for (int i = tid; i < HEADS * CH; i += 256) atts[i] = att[i];
  for (int j = tid; j < EL; j += 256) {
    if (j < EPG) {
      srcs[j] = ei[b * EPG + j] - (int)nb;
      dsts[j] = ei[E0 + b * EPG + j] - (int)nb;
    } else {
      srcs[j] = j - EPG;
      dsts[j] = j - EPG;
    }
  }
  __syncthreads();

  // e[j,h] = att[h] . leaky_relu(xl[src] + xr[dst])
  for (int t = tid; t < EL * HEADS; t += 256) {
    int j = t >> 2, h = t & 3;
    const float* pl = &xls[srcs[j] * HC + h * CH];
    const float* pr = &xrs[dsts[j] * HC + h * CH];
    const float* pa = &atts[h * CH];
    float acc = 0.f;
    for (int c = 0; c < CH; c++) {
      float v = pl[c] + pr[c];
      v = v > 0.f ? v : 0.2f * v;
      acc += v * pa[c];
    }
    es[t] = acc;
  }
  __syncthreads();

  // segment max + sum(exp) per (node, head)
  for (int t = tid; t < NUM_NODES * HEADS; t += 256) {
    int n = t >> 2, h = t & 3;
    float m = -3.4e38f;
    for (int j = 0; j < EL; j++)
      if (dsts[j] == n) m = fmaxf(m, es[j * HEADS + h]);
    float s = 0.f;
    for (int j = 0; j < EL; j++)
      if (dsts[j] == n) s += __expf(es[j * HEADS + h] - m);
    ms[t] = m;
    ss[t] = s;
  }
  __syncthreads();

  // alpha + aggregation (LDS ds_add_f32 atomics)
  for (int t = tid; t < EL * HEADS; t += 256) {
    int j = t >> 2, h = t & 3;
    int dn = dsts[j];
    float a = __expf(es[t] - ms[dn * HEADS + h]) / ss[dn * HEADS + h];
    size_t ge = (j < EPG) ? ((size_t)b * EPG + j)
                          : ((size_t)E0 + nb + (size_t)(j - EPG));
    alpha_out[ge * HEADS + h] = a;
    const float* pl = &xls[srcs[j] * HC + h * CH];
    float* pg = &aggs[dn * HC + h * CH];
    for (int c = 0; c < CH; c++) atomicAdd(&pg[c], a * pl[c]);
  }
  __syncthreads();

  // h = relu(agg + bias); emit [32, 256] with rows 30,31 zeroed
  for (int i = tid; i < LPAD * HC; i += 256) {
    int t = i >> 8, c = i & 255;
    float v = 0.f;
    if (t < NUM_NODES) v = fmaxf(aggs[t * HC + c] + gbias[c], 0.f);
    h_f32[(size_t)b * LPAD * HC + i] = v;
    h_bf[(size_t)b * LPAD * HC + i] = (bf16)v;
  }
}

// ---------------------------------------------------------------------------
// TCN block: conv1d(K=3, dilation DIL, same-pad) as 3 shifted WMMA GEMMs,
// optional fused 1x1 downsample GEMM, then fused LayerNorm-over-time + ReLU +
// residual. One block per graph. All TPW col-tiles of a wave share one
// row-tile, so A fragments are loaded once per (tap,k0) and the inner tile
// loop issues back-to-back WMMAs on INDEPENDENT accumulators (no hazard NOPs).
// ---------------------------------------------------------------------------
template <int CIN, int COUT, int DIL, bool HAS_DS, bool PACK_FC, bool WRITE_SEQ>
__global__ __launch_bounds__(512) void k_tcn(
    const bf16* __restrict__ xin,      // [B, 32, CIN] bf16
    const float* __restrict__ resid,   // [B, 32, CIN] fp32 (used iff !HAS_DS)
    const bf16* __restrict__ Wc,       // [3][COUT][CIN] bf16
    const float* __restrict__ bc,      // [COUT]
    const float* __restrict__ lnw, const float* __restrict__ lnb,  // [30]
    const bf16* __restrict__ Wd,       // [COUT][CIN] bf16 (1x1)
    const float* __restrict__ bd,      // [COUT]
    bf16* __restrict__ out,            // [B, 32, COUT] bf16
    bf16* __restrict__ zpack) {        // [B, COUT*30] bf16, K-order c*30+t
  __shared__ bf16 xs[LPAD * CIN];
  __shared__ float ys[LPAD * COUT];
  __shared__ float dss[HAS_DS ? (LPAD * COUT) : 16];

  const int b = blockIdx.x;
  const int tid = threadIdx.x;
  const int l = tid & 31;
  const int w = tid >> 5;
  const size_t xbase = (size_t)b * LPAD * CIN;

  for (int i = tid; i < LPAD * CIN; i += 512) xs[i] = xin[xbase + i];
  __syncthreads();

  const int NTC = COUT / 16;  // col tiles
  const int TPW = (2 * NTC) / 16;  // tiles per wave (16 waves, 2 row tiles)
  // All tiles of this wave share the row tile (NTC % TPW == 0 for all configs)
  const int rt = (w * TPW) / NTC;
  const int ctb = (w * TPW) % NTC;

  v8f acc[TPW];
  v8f accd[HAS_DS ? TPW : 1];
#pragma unroll
  for (int ti = 0; ti < TPW; ti++) acc[ti] = (v8f){};
  if (HAS_DS) {
#pragma unroll
    for (int ti = 0; ti < TPW; ti++) accd[ti] = (v8f){};
  }

#pragma unroll
  for (int tap = 0; tap < 3; tap++) {
    const bf16* Wt = Wc + (size_t)tap * COUT * CIN;
    int m = rt * 16 + (l & 15) + (tap - 1) * DIL;  // shifted source row
    bool ok = (m >= 0) && (m < NUM_NODES);         // zero pad outside [0,30)
#pragma unroll
    for (int k0 = 0; k0 < CIN; k0 += 32) {
      v16bf a = ok ? load_a(&xs[m * CIN + ((l >> 4) << 3) + k0]) : zero16();
#pragma unroll
      for (int ti = 0; ti < TPW; ti++) {
        v16bf bb = load_b16(Wt, CIN, (ctb + ti) * 16, k0);
        acc[ti] = wmma_bf16(a, bb, acc[ti]);
      }
      if (HAS_DS && tap == 1) {
#pragma unroll
        for (int ti = 0; ti < TPW; ti++) {
          v16bf bd2 = load_b16(Wd, CIN, (ctb + ti) * 16, k0);
          accd[ti] = wmma_bf16(a, bd2, accd[ti]);
        }
      }
    }
  }

#pragma unroll
  for (int ti = 0; ti < TPW; ti++) {
    int col = (ctb + ti) * 16 + (l & 15);
    float cb = bc[col];
#pragma unroll
    for (int i = 0; i < 8; i++) {
      int row = rt * 16 + i + ((l >> 4) << 3);
      ys[row * COUT + col] = acc[ti][i] + cb;
    }
    if (HAS_DS) {
      float db = bd[col];
#pragma unroll
      for (int i = 0; i < 8; i++) {
        int row = rt * 16 + i + ((l >> 4) << 3);
        dss[row * COUT + col] = accd[ti][i] + db;
      }
    }
  }
  __syncthreads();

  // LayerNorm over t (30) per channel, ReLU, + residual; write padded rows = 0
  for (int co = tid; co < COUT; co += 512) {
    float mu = 0.f;
    for (int t = 0; t < NUM_NODES; t++) mu += ys[t * COUT + co];
    mu *= (1.f / NUM_NODES);
    float var = 0.f;
    for (int t = 0; t < NUM_NODES; t++) {
      float d = ys[t * COUT + co] - mu;
      var += d * d;
    }
    var *= (1.f / NUM_NODES);
    float rs = rsqrtf(var + 1e-5f);
    for (int t = 0; t < LPAD; t++) {
      float v = 0.f;
      if (t < NUM_NODES) {
        float n = (ys[t * COUT + co] - mu) * rs * lnw[t] + lnb[t];
        float r = fmaxf(n, 0.f);
        v = r + (HAS_DS ? dss[t * COUT + co] : resid[xbase + t * CIN + co]);
      }
      if (WRITE_SEQ) out[(size_t)b * LPAD * COUT + t * COUT + co] = (bf16)v;
      if (PACK_FC && t < NUM_NODES)
        zpack[(size_t)b * (COUT * NUM_NODES) + co * NUM_NODES + t] = (bf16)v;
    }
  }
}

// ---------------------------------------------------------------------------
// Dense FC GEMM: one wave per 16x16 tile, K streamed with TWO interleaved
// accumulators (even/odd 32-chunks) so consecutive WMMAs are independent.
// ---------------------------------------------------------------------------
template <bool RELU, bool OUT_BF>
__global__ __launch_bounds__(32) void k_fc(
    const bf16* __restrict__ A, const bf16* __restrict__ W,
    const float* __restrict__ bias, bf16* __restrict__ obf,
    float* __restrict__ of32, int K, int N) {
  int l = threadIdx.x & 31;
  int mt = blockIdx.x, nt = blockIdx.y;
  v8f acc0 = {}, acc1 = {};
  const bf16* arow = A + (size_t)(mt * 16 + (l & 15)) * K + ((l >> 4) << 3);
  for (int k0 = 0; k0 < K; k0 += 64) {  // K % 64 == 0 for all layers
    __builtin_prefetch(arow + k0 + 512, 0, 1);  // global_prefetch_b8
    v16bf a0 = load_a(arow + k0);
    v16bf b0 = load_b16(W, K, nt * 16, k0);
    v16bf a1 = load_a(arow + k0 + 32);
    v16bf b1 = load_b16(W, K, nt * 16, k0 + 32);
    acc0 = wmma_bf16(a0, b0, acc0);
    acc1 = wmma_bf16(a1, b1, acc1);
  }
  v8f acc = acc0 + acc1;
  int col = nt * 16 + (l & 15);
  float bv = bias[col];
#pragma unroll
  for (int i = 0; i < 8; i++) {
    int row = mt * 16 + i + ((l >> 4) << 3);
    float v = acc[i] + bv;
    if (RELU) v = fmaxf(v, 0.f);
    if (OUT_BF)
      obf[(size_t)row * N + col] = (bf16)v;
    else
      of32[(size_t)row * N + col] = v;
  }
}

// ei output: [2, ETOT] int32 bits into d_out tail (edges then self loops)
__global__ void k_ei(const int* __restrict__ ei, int* __restrict__ out) {
  int i = blockIdx.x * blockDim.x + threadIdx.x;
  if (i >= ETOT) return;
  int s, d;
  if (i < E0) {
    s = ei[i];
    d = ei[E0 + i];
  } else {
    s = i - E0;
    d = s;
  }
  out[i] = s;
  out[ETOT + i] = d;
}

// ---------------------------------------------------------------------------
extern "C" void kernel_launch(void* const* d_in, const int* in_sizes, int n_in,
                              void* d_out, int out_size, void* d_ws,
                              size_t ws_size, hipStream_t stream) {
  const float* x      = (const float*)d_in[0];
  const float* gWl    = (const float*)d_in[1];
  const float* gbl    = (const float*)d_in[2];
  const float* gWr    = (const float*)d_in[3];
  const float* gbr    = (const float*)d_in[4];
  const float* gatt   = (const float*)d_in[5];
  const float* gbias  = (const float*)d_in[6];
  const float* t0w    = (const float*)d_in[7];
  const float* t0b    = (const float*)d_in[8];
  const float* ln0w   = (const float*)d_in[9];
  const float* ln0b   = (const float*)d_in[10];
  const float* t1w    = (const float*)d_in[11];
  const float* t1b    = (const float*)d_in[12];
  const float* ln1w   = (const float*)d_in[13];
  const float* ln1b   = (const float*)d_in[14];
  const float* d1w    = (const float*)d_in[15];
  const float* d1b    = (const float*)d_in[16];
  const float* t2w    = (const float*)d_in[17];
  const float* t2b    = (const float*)d_in[18];
  const float* ln2w   = (const float*)d_in[19];
  const float* ln2b   = (const float*)d_in[20];
  const float* ds2w   = (const float*)d_in[21];
  const float* d2b    = (const float*)d_in[22];
  const float* f1w    = (const float*)d_in[23];
  const float* f1b    = (const float*)d_in[24];
  const float* f2w    = (const float*)d_in[25];
  const float* f2b    = (const float*)d_in[26];
  const float* f3w    = (const float*)d_in[27];
  const float* f3b    = (const float*)d_in[28];
  const int*   eidx   = (const int*)d_in[29];

  // workspace carve-out (256B aligned)
  char* base = (char*)d_ws;
  size_t off = 0;
  auto take = [&](size_t bytes) -> char* {
    char* p = base + off;
    off += (bytes + 255) & ~(size_t)255;
    return p;
  };
  bf16* x_bf  = (bf16*)take((size_t)NN * F_IN * 2);
  bf16* Wl_bf = (bf16*)take((size_t)HC * F_IN * 2);
  bf16* Wr_bf = (bf16*)take((size_t)HC * F_IN * 2);
  float* xl_f = (float*)take((size_t)NN * HC * 4);
  float* xr_f = (float*)take((size_t)NN * HC * 4);
  float* h_f  = (float*)take((size_t)BG * LPAD * HC * 4);
  bf16* h_bf  = (bf16*)take((size_t)BG * LPAD * HC * 2);
  bf16* w0_bf = (bf16*)take((size_t)3 * 256 * 256 * 2);
  bf16* w1_bf = (bf16*)take((size_t)3 * 512 * 256 * 2);
  bf16* dw1_bf= (bf16*)take((size_t)512 * 256 * 2);
  bf16* w2_bf = (bf16*)take((size_t)3 * 256 * 512 * 2);
  bf16* dw2_bf= (bf16*)take((size_t)256 * 512 * 2);
  bf16* y0_bf = (bf16*)take((size_t)BG * LPAD * 256 * 2);
  bf16* y1_bf = (bf16*)take((size_t)BG * LPAD * 512 * 2);
  bf16* z_bf  = (bf16*)take((size_t)BG * 7680 * 2);
  bf16* f1_bf = (bf16*)take((size_t)512 * 7680 * 2);
  bf16* f2_bf = (bf16*)take((size_t)256 * 512 * 2);
  bf16* f3_bf = (bf16*)take((size_t)144 * 256 * 2);
  bf16* z1_bf = (bf16*)take((size_t)BG * 512 * 2);
  bf16* z2_bf = (bf16*)take((size_t)BG * 256 * 2);

  auto cvt = [&](const float* s, bf16* d, int n) {
    k_f2bf<<<(n + 255) / 256, 256, 0, stream>>>(s, d, n);
  };
  cvt(x, x_bf, NN * F_IN);
  cvt(gWl, Wl_bf, HC * F_IN);
  cvt(gWr, Wr_bf, HC * F_IN);
  k_conv_w<<<(256 * 256 * 3 + 255) / 256, 256, 0, stream>>>(t0w, w0_bf, 256, 256);
  k_conv_w<<<(512 * 256 * 3 + 255) / 256, 256, 0, stream>>>(t1w, w1_bf, 512, 256);
  k_conv_w<<<(256 * 512 * 3 + 255) / 256, 256, 0, stream>>>(t2w, w2_bf, 256, 512);
  cvt(d1w, dw1_bf, 512 * 256);
  cvt(ds2w, dw2_bf, 256 * 512);
  cvt(f1w, f1_bf, 512 * 7680);
  cvt(f2w, f2_bf, 256 * 512);
  cvt(f3w, f3_bf, 144 * 256);

  // GAT transforms (WMMA GEMMs)
  k_xform<<<dim3(NN / 16, 2, 2), 256, 0, stream>>>(x_bf, Wl_bf, Wr_bf, gbl, gbr,
                                                   xl_f, xr_f);

  // GAT attention + aggregation (per-graph block)
  float* alpha_ptr = (float*)d_out + OUT_ELEMS;
  k_gat<<<BG, 256, 0, stream>>>(xl_f, xr_f, eidx, gatt, gbias, alpha_ptr, h_f,
                                h_bf);

  // TCN stack (per-graph block, WMMA conv + fused LN/ReLU/residual)
  k_tcn<256, 256, 1, false, false, true><<<BG, 512, 0, stream>>>(
      h_bf, h_f, w0_bf, t0b, ln0w, ln0b, (const bf16*)nullptr,
      (const float*)nullptr, y0_bf, (bf16*)nullptr);
  k_tcn<256, 512, 2, true, false, true><<<BG, 512, 0, stream>>>(
      y0_bf, (const float*)nullptr, w1_bf, t1b, ln1w, ln1b, dw1_bf, d1b, y1_bf,
      (bf16*)nullptr);
  k_tcn<512, 256, 4, true, true, false><<<BG, 512, 0, stream>>>(
      y1_bf, (const float*)nullptr, w2_bf, t2b, ln2w, ln2b, dw2_bf, d2b,
      (bf16*)nullptr, z_bf);

  // FC head
  k_fc<true, true><<<dim3(BG / 16, 512 / 16), 32, 0, stream>>>(
      z_bf, f1_bf, f1b, z1_bf, (float*)nullptr, 7680, 512);
  k_fc<true, true><<<dim3(BG / 16, 256 / 16), 32, 0, stream>>>(
      z1_bf, f2_bf, f2b, z2_bf, (float*)nullptr, 512, 256);
  k_fc<false, false><<<dim3(BG / 16, 144 / 16), 32, 0, stream>>>(
      z2_bf, f3_bf, f3b, (bf16*)nullptr, (float*)d_out, 256, 144);

  // edge_index output (int bits into fp32 d_out tail)
  int* ei_out = (int*)((float*)d_out + OUT_ELEMS + ALPHA_ELEMS);
  k_ei<<<(ETOT + 255) / 256, 256, 0, stream>>>(eidx, ei_out);
}
